// GNN_26139170964315
// MI455X (gfx1250) — matrix operand order repaired and checked by reading, not compiled, not run on hardware
//
#include <hip/hip_runtime.h>
#include <hip/hip_bf16.h>

// GRU scan for MI455X (gfx1250), wave32 + WMMA bf16.
//  - Kernel 1: swizzle w_ih/w_hh (f32, GATE x H) -> bf16 WMMA B-fragments in d_ws.
//  - Kernel 2: persistent scan. 16 workgroups, each owns a 16-row batch tile;
//    h tile lives in LDS for all T=512 steps. 8 waves split GATE; r/i gates
//    accumulate x-GEMM and h-GEMM into one accumulator; n gate keeps separate
//    x/h accumulators (resetgate scales h_n).
//  - Weight streaming: an opaque per-step SGPR offset (empty asm, "+s") keeps
//    the L2-resident fragment loads inside the t-loop (no LICM -> no spills)
//    while preserving the global address space, so they lower to
//    global_load_b128 (LOADcnt only) instead of flat_load (LOADcnt+DScnt).

#define B_    256
#define T_    512
#define H_    256
#define NT_   48          // GATE/16 column tiles
#define KT_   8           // H/32 K tiles
#define FRAGE 512         // elements per 32x16 bf16 B-fragment
#define PERW  (NT_ * KT_ * FRAGE)
#define WPT   16          // batch rows per workgroup
#define LDSB  264         // bf16 LDS row stride (pad: conflict-free b128)
#define LDSF  260         // f32  LDS row stride

typedef __attribute__((ext_vector_type(16))) __bf16 v16bf;
typedef __attribute__((ext_vector_type(8)))  float  v8f;

__device__ __forceinline__ v8f wmma_bf16(v16bf a, v16bf b, v8f c) {
  // 8 args: (neg_a, A, neg_b, B, c_mod, C, reuse_a, reuse_b)
  return __builtin_amdgcn_wmma_f32_16x16x32_bf16(false, a, false, b,
                                                 (short)0, c, false, false);
}

__device__ __forceinline__ float sigm_f(float x) {
  return 1.0f / (1.0f + __expf(-x));
}
__device__ __forceinline__ float tanh_f(float x) {
  return 1.0f - 2.0f / (__expf(2.0f * x) + 1.0f);   // tanh via one exp
}

// ---------------------------------------------------------------------------
// Kernel 1: f32 weights -> bf16 B-fragments.
// Fragment (nt, kt) holds B[k][n] = W[nt*16+n][kt*32+k] in this code's contiguous-k (same as the A reads)
// 32x16 bf16 B layout: lane l -> n=l&15, kbase=(l>>4)*16; element e (0..15)
// of the lane's v16bf -> K = kbase + e.  Stored lane-contiguous (32B/lane)
// so the scan kernel loads each fragment as 2x global_load_b128 per lane.
// ---------------------------------------------------------------------------
__global__ void swizzle_weights(const float* __restrict__ w_ih,
                                const float* __restrict__ w_hh,
                                __bf16* __restrict__ frags) {
  int idx = blockIdx.x * blockDim.x + threadIdx.x;   // 0 .. 2*PERW-1
  int w   = idx / PERW;
  int rem = idx - w * PERW;
  int fi  = rem >> 9;            // nt*KT_ + kt
  int e2  = rem & (FRAGE - 1);   // lane*16 + e
  int nt  = fi >> 3;
  int kt  = fi & 7;
  int lane = e2 >> 4;
  int e    = e2 & 15;
  int g = nt * 16 + (lane & 15);
  int k = kt * 32 + ((lane >> 4) << 4) + e;
  const float* W = (w == 0) ? w_ih : w_hh;
  frags[idx] = (__bf16)W[g * H_ + k];
}

// ---------------------------------------------------------------------------
// Kernel 2: persistent GRU scan, one workgroup per 16-row batch tile.
// ---------------------------------------------------------------------------
__global__ __launch_bounds__(256)
void gru_scan(const float* __restrict__ hidden,
              const float* __restrict__ gii,
              const float* __restrict__ b_ih,
              const float* __restrict__ b_hh,
              const float* __restrict__ h0,
              const __bf16* __restrict__ frags,
              float* __restrict__ out) {
  __shared__ __bf16 xb[WPT * LDSB];   // x(t) tile, bf16
  __shared__ __bf16 hb[WPT * LDSB];   // h     tile, bf16 (WMMA A side)
  __shared__ float  hf[WPT * LDSF];   // h     tile, f32  (exact update)
  __shared__ float  gs[WPT];          // g_item_interest column at t

  const int tid  = threadIdx.x;
  const int lane = tid & 31;
  const int wave = tid >> 5;          // 0..7 (wave32)
  const int b0   = blockIdx.x * WPT;

  // ---- stage h0 into hf / hb ----
  {
    int r  = tid >> 4;
    int c0 = (tid & 15) << 4;
    const float* src = h0 + (size_t)(b0 + r) * H_ + c0;
#pragma unroll
    for (int i = 0; i < 16; i += 4) {
      float4 v = *(const float4*)(src + i);
      hf[r * LDSF + c0 + i + 0] = v.x;  hb[r * LDSB + c0 + i + 0] = (__bf16)v.x;
      hf[r * LDSF + c0 + i + 1] = v.y;  hb[r * LDSB + c0 + i + 1] = (__bf16)v.y;
      hf[r * LDSF + c0 + i + 2] = v.z;  hb[r * LDSB + c0 + i + 2] = (__bf16)v.z;
      hf[r * LDSF + c0 + i + 3] = v.w;  hb[r * LDSB + c0 + i + 3] = (__bf16)v.w;
    }
  }

  // ---- per-lane biases for this wave's two gate-column tiles ----
  const int nloc = lane & 15;
  float biasR[2], biasI[2], biasIN[2], biasHN[2];
#pragma unroll
  for (int j = 0; j < 2; ++j) {
    int col = (wave * 2 + j) * 16 + nloc;          // column in [0, H)
    biasR[j]  = b_ih[col]          + b_hh[col];
    biasI[j]  = b_ih[H_ + col]     + b_hh[H_ + col];
    biasIN[j] = b_ih[2 * H_ + col];
    biasHN[j] = b_hh[2 * H_ + col];
  }

  __syncthreads();

  for (int t = 0; t < T_; ++t) {
    // Opaque per-step scalar offset: loads below depend on it, so they cannot
    // be hoisted out of the t-loop (no spill), yet the base pointer keeps its
    // global address space -> global_load_b128 (saddr + voffset form).
    unsigned zoff = 0;
    asm volatile("" : "+s"(zoff));
    const __bf16* wihF = frags + zoff;
    const __bf16* whhF = frags + PERW + zoff;

    // ---- stage x(t) tile and g column ----
    {
      int r  = tid >> 4;
      int c0 = (tid & 15) << 4;
      const float* src = hidden + ((size_t)(b0 + r) * T_ + t) * H_ + c0;
#pragma unroll
      for (int i = 0; i < 16; i += 4) {
        float4 v = *(const float4*)(src + i);
        xb[r * LDSB + c0 + i + 0] = (__bf16)v.x;
        xb[r * LDSB + c0 + i + 1] = (__bf16)v.y;
        xb[r * LDSB + c0 + i + 2] = (__bf16)v.z;
        xb[r * LDSB + c0 + i + 3] = (__bf16)v.w;
      }
      if (t + 1 < T_) __builtin_prefetch(src + H_, 0, 1);   // next step's row
      if (tid < WPT) gs[tid] = gii[(size_t)(b0 + tid) * T_ + t];
    }
    __syncthreads();

    // ---- accumulators: r/i fused (x+h), n split ----
    v8f accR[2], accI[2], accIN[2], accHN[2];
#pragma unroll
    for (int j = 0; j < 2; ++j)
#pragma unroll
      for (int i = 0; i < 8; ++i) {
        accR[j][i] = 0.f; accI[j][i] = 0.f; accIN[j][i] = 0.f; accHN[j][i] = 0.f;
      }

#pragma unroll
    for (int kt = 0; kt < KT_; ++kt) {
      // A fragments (16x32 bf16): two aligned 16B LDS reads each.
      v16bf ax, ah;
      {
        int m    = lane & 15;
        int kb   = (lane >> 4) << 4;
        int base = m * LDSB + kt * 32 + kb;
#pragma unroll
        for (int i = 0; i < 8; ++i) {
          ax[i]     = xb[base + i];
          ax[8 + i] = xb[base + 8 + i];
          ah[i]     = hb[base + i];
          ah[8 + i] = hb[base + 8 + i];
        }
      }
      // Load all 12 B fragments for this k-step (coalesced 2x b128 per lane),
      // then issue the 12 WMMAs.
      v16bf bR1[2], bR2[2], bI1[2], bI2[2], bN1[2], bN2[2];
#pragma unroll
      for (int j = 0; j < 2; ++j) {
        const int jt = wave * 2 + j;
        const int lo = lane * 16;
        bR1[j] = *(const v16bf*)(wihF + (((jt     ) * KT_ + kt) << 9) + lo);
        bR2[j] = *(const v16bf*)(whhF + (((jt     ) * KT_ + kt) << 9) + lo);
        bI1[j] = *(const v16bf*)(wihF + (((16 + jt) * KT_ + kt) << 9) + lo);
        bI2[j] = *(const v16bf*)(whhF + (((16 + jt) * KT_ + kt) << 9) + lo);
        bN1[j] = *(const v16bf*)(wihF + (((32 + jt) * KT_ + kt) << 9) + lo);
        bN2[j] = *(const v16bf*)(whhF + (((32 + jt) * KT_ + kt) << 9) + lo);
      }
#pragma unroll
      for (int j = 0; j < 2; ++j) {
        accR[j]  = wmma_bf16(ax, bR1[j], accR[j]);
        accR[j]  = wmma_bf16(ah, bR2[j], accR[j]);
        accI[j]  = wmma_bf16(ax, bI1[j], accI[j]);
        accI[j]  = wmma_bf16(ah, bI2[j], accI[j]);
        accIN[j] = wmma_bf16(ax, bN1[j], accIN[j]);
        accHN[j] = wmma_bf16(ah, bN2[j], accHN[j]);
      }
    }

    // ---- gates + h update (C/D layout: VGPR vr, lane -> M = vr + (lane>>4)*8,
    //      N = lane&15) ----
    float hy[2][8];
#pragma unroll
    for (int j = 0; j < 2; ++j) {
      int col = (wave * 2 + j) * 16 + nloc;
#pragma unroll
      for (int vr = 0; vr < 8; ++vr) {
        int m = vr + ((lane >> 4) << 3);
        float rg = sigm_f(accR[j][vr] + biasR[j]);
        float ig = sigm_f(accI[j][vr] + biasI[j]);
        float ng = tanh_f(accIN[j][vr] + biasIN[j] +
                          rg * (accHN[j][vr] + biasHN[j]));
        float gv = gs[m];
        float cg = (gv > 0.01f) ? gv * ig : 0.0f;
        float ho = hf[m * LDSF + col];
        hy[j][vr] = (1.0f - cg) * ho + cg * ng;
      }
    }
    __syncthreads();   // all waves done reading h (bf16 + f32)
#pragma unroll
    for (int j = 0; j < 2; ++j) {
      int col = (wave * 2 + j) * 16 + nloc;
#pragma unroll
      for (int vr = 0; vr < 8; ++vr) {
        int m = vr + ((lane >> 4) << 3);
        hf[m * LDSF + col] = hy[j][vr];
        hb[m * LDSB + col] = (__bf16)hy[j][vr];
      }
    }
    __syncthreads();   // h ready for next step
  }

  // ---- write h_final tile ----
  {
    int r  = tid >> 4;
    int c0 = (tid & 15) << 4;
    float* dst = out + (size_t)(b0 + r) * H_ + c0;
#pragma unroll
    for (int i = 0; i < 16; i += 4) {
      float4 v;
      v.x = hf[r * LDSF + c0 + i + 0];
      v.y = hf[r * LDSF + c0 + i + 1];
      v.z = hf[r * LDSF + c0 + i + 2];
      v.w = hf[r * LDSF + c0 + i + 3];
      *(float4*)(dst + i) = v;
    }
  }
}

extern "C" void kernel_launch(void* const* d_in, const int* in_sizes, int n_in,
                              void* d_out, int out_size, void* d_ws, size_t ws_size,
                              hipStream_t stream) {
  const float* hidden = (const float*)d_in[0];
  const float* gii    = (const float*)d_in[1];
  const float* w_ih   = (const float*)d_in[2];
  const float* w_hh   = (const float*)d_in[3];
  const float* b_ih   = (const float*)d_in[4];
  const float* b_hh   = (const float*)d_in[5];
  const float* h0     = (const float*)d_in[6];
  float* out = (float*)d_out;

  __bf16* frags = (__bf16*)d_ws;                 // 2 * PERW bf16 = 768 KB
  const int total = 2 * PERW;

  swizzle_weights<<<total / 256, 256, 0, stream>>>(w_ih, w_hh, frags);
  gru_scan<<<B_ / WPT, 256, 0, stream>>>(hidden, gii, b_ih, b_hh, h0, frags, out);
}
